// SelfAttention_54820962566747
// MI455X (gfx1250) — matrix operand order repaired and checked by reading, not compiled, not run on hardware
//
#include <hip/hip_runtime.h>
#include <hip/hip_bf16.h>

typedef __attribute__((ext_vector_type(16))) __bf16 v16bf;
typedef __attribute__((ext_vector_type(8)))  float  v8f;
typedef __attribute__((ext_vector_type(4)))  unsigned int   u32x4;
typedef __attribute__((ext_vector_type(8)))  unsigned short u16x8;
typedef __attribute__((ext_vector_type(4)))  unsigned short u16x4;
typedef __attribute__((ext_vector_type(4)))  float          f32x4;

constexpr int TT  = 128;   // sequence length
constexpr int CC  = 192;   // embed / head dim
constexpr int SRX = 200;   // row stride (elems) for xs/qs/ks (400B: 16B-mult, bank-friendly)
constexpr int SVS = 136;   // row stride (elems) for vs = V^T [h][s]
constexpr int SRP = 136;   // row stride (elems) for ps (probabilities [t][s])

union BV { u32x4 u[2]; v16bf v; };

static __device__ __forceinline__ v8f zero8() {
  v8f z = {0.f, 0.f, 0.f, 0.f, 0.f, 0.f, 0.f, 0.f};
  return z;
}

static __device__ __forceinline__ v8f wmma_bf16(v16bf a, v16bf b, v8f c) {
  // D(f32 16x16) = A(bf16 16x32) * B(bf16 32x16) + C
  return __builtin_amdgcn_wmma_f32_16x16x32_bf16(false, a, false, b,
                                                 (short)0, c, false, false);
}

// A tile (16x32, MxK) from row-major [row][col] bf16 buffer.
// ISA A layout: lane<16: v0-3 = K0..7, v4-7 = K16..23 ; lane>=16: K8..15 / K24..31
static __device__ __forceinline__ v16bf load_tile_A(const __bf16* base, int stride,
                                                    int row0, int col0, int lr, int hf) {
  BV r;
  const __bf16* p0 = base + (row0 + lr) * stride + col0 + hf * 8;
  r.u[0] = *(const u32x4*)(p0);        // 8 bf16
  r.u[1] = *(const u32x4*)(p0 + 16);   // 8 bf16
  return r.v;
}

// B tile (32x16, KxN) from [N][K] row-major bf16 buffer.
// ISA B layout: lane<16 (n=lane): v0-3 = K0..7, v4-7 = K16..23 ; lane>=16: K8..15 / K24..31
static __device__ __forceinline__ v16bf load_tile_B(const __bf16* base, int stride,
                                                    int n0, int k0, int lr, int hf) {
  BV r;
  const __bf16* p0 = base + (n0 + lr) * stride + k0 + hf * 8;
  r.u[0] = *(const u32x4*)(p0);
  r.u[1] = *(const u32x4*)(p0 + 16);
  return r.v;
}

// Pack 8 f32 accumulator values -> 8 bf16 -> one 16B store (contiguous)
static __device__ __forceinline__ void store_pk8(__bf16* dst, v8f a) {
  u16x8 p;
#pragma unroll
  for (int r = 0; r < 8; ++r)
    p[r] = __builtin_bit_cast(unsigned short, (__bf16)a[r]);
  *(u16x8*)dst = p;
}

// ---- weight fp32 -> bf16 prep (runs once per launch; tiny, L2-resident after) ----
__global__ void cvt_weights_bf16(const float* __restrict__ wk,
                                 const float* __restrict__ wq,
                                 const float* __restrict__ wv,
                                 __bf16* __restrict__ o) {
  const int N = CC * CC;
  int i = blockIdx.x * blockDim.x + threadIdx.x;
  if (i < N)            o[i] = (__bf16)wk[i];
  else if (i < 2 * N)   o[i] = (__bf16)wq[i - N];
  else if (i < 3 * N)   o[i] = (__bf16)wv[i - 2 * N];
}

// ---- fused causal self-attention: one workgroup (8 waves) per batch element ----
// LDS = 151KB -> 2 resident blocks per 320KB WGP (4 waves/SIMD).
__global__ __launch_bounds__(256)
void attn_fused_kernel(const float* __restrict__ x,
                       const __bf16* __restrict__ wkb,
                       const __bf16* __restrict__ wqb,
                       const __bf16* __restrict__ wvb,
                       float* __restrict__ out) {
  extern __shared__ __bf16 smem[];
  // buf0 is triple-purpose: xs (x bf16) -> qs (Q [t][h]) -> ps (P [t][s]).
  //  xs->qs: wave w's Q stores hit only its own 16 rows, which it already
  //          consumed into registers (per-wave LDS ops are in-order).
  //  qs->ps: guarded by a barrier after all Bq hoists (ps rows cross waves).
  __bf16* xs = smem;                 // [128][SRX]
  __bf16* qs = smem;                 // [128][SRX]  ([t][h]: B layout for S^T=K·Q^T)
  __bf16* ps = smem;                 // [128][SRP]  ([t][s]: B layout for O^T=V^T·P^T)
  __bf16* ks = smem + TT * SRX;      // [128][SRX]  ([s][h]: A layout for S^T)
  __bf16* vs = ks + TT * SRX;        // [192][SVS]  ([h][s]: A layout for O^T)

  const int b    = blockIdx.x;
  const int tid  = threadIdx.x;
  const int lane = tid & 31;
  const int wave = tid >> 5;         // 8 waves, wave w owns rows 16w..16w+15
  const int lr   = lane & 15;
  const int hf   = lane >> 4;
  const int t0   = wave * 16;

  // ---- 1) x -> bf16 -> LDS (float4 in, packed 4x bf16 out) ----
  const float* xb = x + (size_t)b * TT * CC;
  for (int i = tid; i < (TT * CC) / 4; i += 256) {
    int t = i / (CC / 4), c4 = (i % (CC / 4)) * 4;
    f32x4 f = *(const f32x4*)(xb + t * CC + c4);
    u16x4 p;
    p.x = __builtin_bit_cast(unsigned short, (__bf16)f.x);
    p.y = __builtin_bit_cast(unsigned short, (__bf16)f.y);
    p.z = __builtin_bit_cast(unsigned short, (__bf16)f.z);
    p.w = __builtin_bit_cast(unsigned short, (__bf16)f.w);
    *(u16x4*)(xs + t * SRX + c4) = p;
  }
  __syncthreads();

  // ---- 2) projections.  Q^T/K^T = W · x^T  (contiguous packed stores into [t][h])
  //                       V       = x · Wv^T (contiguous packed stores into [h][s])
  v16bf Ax[6], Bx[6];
#pragma unroll
  for (int kk = 0; kk < 6; ++kk) {
    Ax[kk] = load_tile_A(xs, SRX, t0, kk * 32, lr, hf);   // x rows t (A side, for V)
    Bx[kk] = load_tile_B(xs, SRX, t0, kk * 32, lr, hf);   // x^T      (B side, for Q/K)
  }

  for (int j = 0; j < 12; ++j) {           // h tiles
    v8f aq = zero8(), ak = zero8(), av = zero8();
#pragma unroll
    for (int kk = 0; kk < 6; ++kk) {       // c tiles (K=32)
      v16bf awq = load_tile_A(wqb, CC, j * 16, kk * 32, lr, hf);  // Wq rows h
      v16bf awk = load_tile_A(wkb, CC, j * 16, kk * 32, lr, hf);  // Wk rows h
      v16bf bwv = load_tile_B(wvb, CC, j * 16, kk * 32, lr, hf);  // Wv^T
      aq = wmma_bf16(awq, Bx[kk], aq);     // Q^T tile: (m=h, n=t)
      ak = wmma_bf16(awk, Bx[kk], ak);     // K^T tile: (m=h, n=t)
      av = wmma_bf16(Ax[kk], bwv, av);     // V tile:   (m=t, n=h)
    }
    // Q^T/K^T D elem (m = r+8hf -> h, n = lr -> t): 8 consecutive h per lane
    store_pk8(qs + (t0 + lr) * SRX + j * 16 + 8 * hf, aq);  // own rows: xs overlay OK
    store_pk8(ks + (t0 + lr) * SRX + j * 16 + 8 * hf, ak);
    // V D elem (m = r+8hf -> t, n = lr -> h): 8 consecutive s per lane in vs[h][s]
    store_pk8(vs + (j * 16 + lr) * SVS + t0 + 8 * hf, av);
  }
  __syncthreads();   // K/V visible to all waves

  // ---- 3) S^T = K · Q^T (per wave: its t columns), causal softmax -> P ----
  v16bf Bq[6];
#pragma unroll
  for (int kk = 0; kk < 6; ++kk)
    Bq[kk] = load_tile_B(qs, SRX, t0, kk * 32, lr, hf);   // Q^T for this wave's rows
  __syncthreads();   // all Bq hoists done before anyone overwrites buf0 with ps

  v8f S[8];
#pragma unroll
  for (int i = 0; i < 8; ++i) S[i] = zero8();

#pragma unroll
  for (int i = 0; i < 8; ++i) {
    if (i <= wave) {                       // wave-uniform: EXEC stays full
      v8f acc = zero8();
#pragma unroll
      for (int kk = 0; kk < 6; ++kk) {
        v16bf ak = load_tile_A(ks, SRX, i * 16, kk * 32, lr, hf); // K rows s
        acc = wmma_bf16(ak, Bq[kk], acc);  // (m = s, n = t)
      }
      S[i] = acc;
    }
  }

  // softmax over s for row t = t0 + lr: this lane holds s = 16i + 8hf + r;
  // the other 8 s-values of each tile live in lane^16 -> one shfl_xor(16).
  const float scale = 0.07216878364870323f;   // 1/sqrt(192)
  float mx = -3.0e38f;
#pragma unroll
  for (int i = 0; i < 8; ++i) {
    if (i <= wave) {
#pragma unroll
      for (int r = 0; r < 8; ++r) {
        float v = S[i][r] * scale;
        if (i == wave && (8 * hf + r) > lr) v = -3.0e38f;   // causal mask (diag tile)
        S[i][r] = v;
        mx = fmaxf(mx, v);
      }
    }
  }
  mx = fmaxf(mx, __shfl_xor(mx, 16, 32));
  float sum = 0.f;
#pragma unroll
  for (int i = 0; i < 8; ++i) {
    if (i <= wave) {
#pragma unroll
      for (int r = 0; r < 8; ++r) {
        float e = __expf(S[i][r] - mx);
        S[i][r] = e;
        sum += e;
      }
    }
  }
  sum += __shfl_xor(sum, 16, 32);
  float inv = 1.0f / sum;
#pragma unroll
  for (int i = 0; i < 8; ++i) {
    if (i <= wave) {
      v8f p = S[i];
#pragma unroll
      for (int r = 0; r < 8; ++r) p[r] *= inv;
      // P[t][s]: t = t0+lr, s = 16i + 8hf + r -> contiguous packed store
      store_pk8(ps + (t0 + lr) * SRP + i * 16 + 8 * hf, p);
    }
  }
  // ps read only by this wave; per-wave LDS ops are in-order -> no barrier needed

  // ---- 4) O^T = V^T · P^T: A = vs[h][s], B = ps[t][s]; D(m=h, n=t) gives each
  //      lane 8 consecutive h for its row t -> 32B vector stores to global. ----
  v16bf Bp[8];
#pragma unroll
  for (int i = 0; i < 8; ++i)
    if (i <= wave) Bp[i] = load_tile_B(ps, SRP, t0, i * 16, lr, hf);  // own rows

  float* ob = out + (size_t)b * TT * CC;
  for (int j = 0; j < 12; ++j) {
    v8f acc = zero8();
#pragma unroll
    for (int i = 0; i < 8; ++i) {
      if (i <= wave) {
        v16bf av = load_tile_A(vs, SVS, j * 16, i * 16, lr, hf);  // V^T rows h
        acc = wmma_bf16(av, Bp[i], acc);   // (m = h, n = t)
      }
    }
    // out[t0+lr][16j + 8hf + 0..7]: two 16B stores per lane
    float* po = ob + (t0 + lr) * CC + j * 16 + 8 * hf;
    f32x4 lo = {acc[0], acc[1], acc[2], acc[3]};
    f32x4 hi = {acc[4], acc[5], acc[6], acc[7]};
    *(f32x4*)(po)     = lo;
    *(f32x4*)(po + 4) = hi;
  }
}

extern "C" void kernel_launch(void* const* d_in, const int* in_sizes, int n_in,
                              void* d_out, int out_size, void* d_ws, size_t ws_size,
                              hipStream_t stream) {
  const float* x  = (const float*)d_in[0];
  const float* Wk = (const float*)d_in[1];
  const float* Wq = (const float*)d_in[2];
  const float* Wv = (const float*)d_in[3];
  float* out = (float*)d_out;

  const int B = in_sizes[0] / (TT * CC);
  const int N = CC * CC;

  __bf16* wbf = (__bf16*)d_ws;   // [wk | wq | wv] bf16, 3*192*192*2 = 216KB

  cvt_weights_bf16<<<(3 * N + 255) / 256, 256, 0, stream>>>(Wk, Wq, Wv, wbf);

  // 2*128*200 + 192*136 elems = 151KB -> two blocks fit in a 320KB WGP
  size_t shmem = (size_t)(2 * TT * SRX + CC * SVS) * sizeof(__bf16);
  hipFuncSetAttribute((const void*)attn_fused_kernel,
                      hipFuncAttributeMaxDynamicSharedMemorySize, (int)shmem);

  attn_fused_kernel<<<B, 256, shmem, stream>>>(x, wbf, wbf + N, wbf + 2 * N, out);
}
